// RegionLoss_3D_61263413510182
// MI455X (gfx1250) — matrix operand order repaired and checked by reading, not compiled, not run on hardware
//
#include <hip/hip_runtime.h>

typedef __attribute__((ext_vector_type(16))) _Float16 v16h;
typedef __attribute__((ext_vector_type(8)))  float    v8f;

#define NROWS  8192              // B * S^3 = 2 * 4096
#define NTILES (NROWS / 16)      // 512 row-tiles of 16

// ---------------------------------------------------------------------------
// Kernel 1: exact 4x4x4 average pool.
// In layout (b,c,z4,y4,x4) fp32; out layout (b, s=z*256+y*16+x, c) fp32.
// One thread per pooled element; float4 loads are fully coalesced across x.
// ---------------------------------------------------------------------------
__global__ void pool_kernel(const float* __restrict__ in, float* __restrict__ pooled) {
    int t = blockIdx.x * blockDim.x + threadIdx.x;   // 262144 threads
    int x = t & 15;
    int y = (t >> 4) & 15;
    int z = (t >> 8) & 15;
    int c = (t >> 12) & 31;
    int b = t >> 17;
    const float* src = in + ((((size_t)(b * 32 + c) * 64 + (size_t)z * 4) * 64
                              + (size_t)y * 4) * 64 + (size_t)x * 4);
    float sum = 0.f;
#pragma unroll
    for (int dz = 0; dz < 4; ++dz) {
#pragma unroll
        for (int dy = 0; dy < 4; ++dy) {
            float4 v = *reinterpret_cast<const float4*>(src + ((size_t)dz * 64 + dy) * 64);
            sum += v.x + v.y + v.z + v.w;
        }
    }
    int s = (z << 8) | (y << 4) | x;
    pooled[((size_t)(b * 4096 + s)) * 32 + c] = sum * (1.0f / 64.0f);
}

// ---------------------------------------------------------------------------
// Kernel 2: per-row L2 normalize (rows of 32 channels), fp32 -> fp16.
// One wave32 per row; lane == channel. scale = 1 / max(||row||, 1e-8).
// ---------------------------------------------------------------------------
__global__ void normalize_kernel(const float* __restrict__ pooled, _Float16* __restrict__ out) {
    int gid  = blockIdx.x * blockDim.x + threadIdx.x;
    int row  = gid >> 5;
    int lane = threadIdx.x & 31;
    float v  = pooled[(size_t)row * 32 + lane];
    float ss = v * v;
#pragma unroll
    for (int off = 16; off > 0; off >>= 1) ss += __shfl_xor(ss, off, 32);
    float scale = 1.0f / fmaxf(sqrtf(ss), 1e-8f);
    out[(size_t)row * 32 + lane] = (_Float16)(v * scale);
}

// ---------------------------------------------------------------------------
// WMMA fragment loaders (rows stored contiguously: 32 halves = 64 B per row).
// Layouts per CDNA5 ISA 7.12.2 (wave32).
// ---------------------------------------------------------------------------
__device__ __forceinline__ v16h load_afrag(const _Float16* base, int row, int halfId) {
    // A (16x32 f16): lane<16 -> K 0..7 then 16..23 ; lane>=16 -> K 8..15 then 24..31
    const char* p = reinterpret_cast<const char*>(base) + (size_t)row * 64 + halfId * 16;
    v16h r;
    reinterpret_cast<float4*>(&r)[0] = *reinterpret_cast<const float4*>(p);
    reinterpret_cast<float4*>(&r)[1] = *reinterpret_cast<const float4*>(p + 32);
    return r;
}
__device__ __forceinline__ v16h load_bfrag(const _Float16* base, int row, int halfId) {
    // B (32x16 f16): lane n<16 -> col n, K 0..7 then 16..23 ; lane 16+n -> col n, K 8..15 then 24..31
    const char* p = reinterpret_cast<const char*>(base) + (size_t)row * 64 + halfId * 16;
    v16h r;
    reinterpret_cast<float4*>(&r)[0] = *reinterpret_cast<const float4*>(p);
    reinterpret_cast<float4*>(&r)[1] = *reinterpret_cast<const float4*>(p + 32);
    return r;
}

#define WMMA_F16(A, B) \
    __builtin_amdgcn_wmma_f32_16x16x32_f16(false, (A), false, (B), (short)0, zero, false, false)

// ---------------------------------------------------------------------------
// Kernel 3: fused pairwise-similarity MSE.
// Block = 2 i-tiles, 8 waves. Wave holds A-frags for (An,Bn) x 2 i-tiles and
// sweeps j-tiles two at a time: 8 independent 16x16x32 WMMA chains per
// iteration fill the f16 WMMA->VALU hazard window (ISA 7.12.1) with real work.
// __launch_bounds__(256,1) lifts the VGPR cap so the ~170-VGPR working set
// stays register-resident (round-2 build spilled the A-frags to scratch).
// Deterministic block reduction -> partial[blockIdx].
// ---------------------------------------------------------------------------
__global__ void __launch_bounds__(256, 1)
pairwise_mse_kernel(const _Float16* __restrict__ An,
                    const _Float16* __restrict__ Bn,
                    float* __restrict__ partial) {
    const int lane   = threadIdx.x & 31;
    const int wave   = threadIdx.x >> 5;      // 0..7
    const int halfId = lane >> 4;
    const int m      = lane & 15;
    const int i0     = blockIdx.x * 2;        // two i-tiles per block

    v16h aA0 = load_afrag(An, i0 * 16 + m, halfId);
    v16h aB0 = load_afrag(Bn, i0 * 16 + m, halfId);
    v16h aA1 = load_afrag(An, (i0 + 1) * 16 + m, halfId);
    v16h aB1 = load_afrag(Bn, (i0 + 1) * 16 + m, halfId);

    v8f s0 = {}, s1 = {};
    const v8f zero = {};

    // Each wave owns j-tile pairs {2w, 2w+1}, {2w+16, 2w+17}, ... (512/16 = 32 iters)
    for (int j = wave * 2; j < NTILES; j += 16) {
        v16h bA0 = load_bfrag(An, j * 16 + m, halfId);
        v16h bB0 = load_bfrag(Bn, j * 16 + m, halfId);
        v16h bA1 = load_bfrag(An, (j + 1) * 16 + m, halfId);
        v16h bB1 = load_bfrag(Bn, (j + 1) * 16 + m, halfId);

        v8f cA00 = WMMA_F16(aA0, bA0);
        v8f cB00 = WMMA_F16(aB0, bB0);
        v8f cA10 = WMMA_F16(aA1, bA0);
        v8f cB10 = WMMA_F16(aB1, bB0);
        v8f cA01 = WMMA_F16(aA0, bA1);
        v8f cB01 = WMMA_F16(aB0, bB1);
        v8f cA11 = WMMA_F16(aA1, bA1);
        v8f cB11 = WMMA_F16(aB1, bB1);

        v8f d0 = cA00 - cB00;
        v8f d1 = cA10 - cB10;
        v8f d2 = cA01 - cB01;
        v8f d3 = cA11 - cB11;
        s0 += d0 * d0;
        s1 += d1 * d1;
        s0 += d2 * d2;
        s1 += d3 * d3;
    }

    v8f st = s0 + s1;
    float tot = st[0] + st[1] + st[2] + st[3] + st[4] + st[5] + st[6] + st[7];
#pragma unroll
    for (int off = 16; off > 0; off >>= 1) tot += __shfl_xor(tot, off, 32);

    __shared__ float sm[8];
    if (lane == 0) sm[wave] = tot;
    __syncthreads();
    if (threadIdx.x == 0) {
        float s = 0.f;
#pragma unroll
        for (int w = 0; w < 8; ++w) s += sm[w];
        partial[blockIdx.x] = s;
    }
}

// ---------------------------------------------------------------------------
// Kernel 4: deterministic final reduction of 256 block partials -> mean.
// ---------------------------------------------------------------------------
__global__ void finalize_kernel(const float* __restrict__ partial, float* __restrict__ out) {
    int lane = threadIdx.x & 31;
    int wave = threadIdx.x >> 5;
    float v = partial[threadIdx.x];
#pragma unroll
    for (int off = 16; off > 0; off >>= 1) v += __shfl_xor(v, off, 32);
    __shared__ float sm[8];
    if (lane == 0) sm[wave] = v;
    __syncthreads();
    if (threadIdx.x == 0) {
        float s = 0.f;
#pragma unroll
        for (int w = 0; w < 8; ++w) s += sm[w];
        out[0] = s * (1.0f / ((float)NROWS * (float)NROWS));
    }
}

extern "C" void kernel_launch(void* const* d_in, const int* in_sizes, int n_in,
                              void* d_out, int out_size, void* d_ws, size_t ws_size,
                              hipStream_t stream) {
    (void)in_sizes; (void)n_in; (void)out_size; (void)ws_size;
    const float* p1 = (const float*)d_in[0];
    const float* p2 = (const float*)d_in[1];

    char* ws = (char*)d_ws;
    _Float16* An      = (_Float16*)(ws);                 // 512 KB
    _Float16* Bn      = (_Float16*)(ws + (512 << 10));   // 512 KB
    float*    pooled  = (float*)(ws + (1 << 20));        // 1 MB scratch (reused)
    float*    partial = (float*)(ws + (2 << 20));        // 256 floats
    float*    out     = (float*)d_out;

    pool_kernel<<<1024, 256, 0, stream>>>(p1, pooled);
    normalize_kernel<<<1024, 256, 0, stream>>>(pooled, An);
    pool_kernel<<<1024, 256, 0, stream>>>(p2, pooled);
    normalize_kernel<<<1024, 256, 0, stream>>>(pooled, Bn);
    pairwise_mse_kernel<<<NTILES / 2, 256, 0, stream>>>(An, Bn, partial);
    finalize_kernel<<<1, 256, 0, stream>>>(partial, out);
}
